// MultiHeadAttention_87651692577155
// MI455X (gfx1250) — compile-verified, matrix operand order checked by reading, not run on hardware
//
#include <hip/hip_runtime.h>
#include <hip/hip_bf16.h>

typedef __bf16 bf16;
typedef bf16  bf16x4 __attribute__((ext_vector_type(4)));
typedef bf16  bf16x8 __attribute__((ext_vector_type(8)));
typedef bf16  v16bf  __attribute__((ext_vector_type(16)));
typedef float v8f    __attribute__((ext_vector_type(8)));
typedef int   v4i    __attribute__((ext_vector_type(4)));

typedef __attribute__((address_space(1))) v4i gv4i;  // global 16B chunk
typedef __attribute__((address_space(3))) v4i lv4i;  // LDS 16B chunk

#define EMBED 1024
#define HEADS 16
#define HEADD 64
#define BATCH 2
#define SEQ   2048
#define MROWS (BATCH * SEQ)      // 4096
#define LOG2E 1.4426950408889634f
#define SM_SCALE 0.125f          // 1/sqrt(64)

#if __has_builtin(__builtin_amdgcn_global_load_async_to_lds_b128)
#define ASYNC_STAGE 1
#else
#define ASYNC_STAGE 0
#endif

// ---- helpers -------------------------------------------------------------

static __device__ inline v16bf load_frag(const bf16* p0, const bf16* p1) {
  bf16x8 lo = *(const bf16x8*)p0;
  bf16x8 hi = *(const bf16x8*)p1;
  return __builtin_shufflevector(lo, hi, 0,1,2,3,4,5,6,7,8,9,10,11,12,13,14,15);
}

static __device__ inline v8f wmma_bf16(v16bf a, v16bf b, v8f c) {
  return __builtin_amdgcn_wmma_f32_16x16x32_bf16(false, a, false, b, (short)0, c,
                                                 false, false);
}

// lanes l <-> l+16 exchange (SWAPX16 swizzle pattern)
static __device__ inline float swz16(float x) {
  return __int_as_float(__builtin_amdgcn_ds_swizzle(__float_as_int(x), 0x401f));
}

// 16-byte global->LDS copy: async DMA (ASYNCcnt) when available
static __device__ inline void stage16(const bf16* g, bf16* l) {
#if ASYNC_STAGE
  __builtin_amdgcn_global_load_async_to_lds_b128((gv4i*)(void*)g, (lv4i*)(void*)l,
                                                 0, 0);
#else
  *(bf16x8*)l = *(const bf16x8*)g;
#endif
}

static __device__ inline void stage_wait() {
#if ASYNC_STAGE
#if __has_builtin(__builtin_amdgcn_s_wait_asynccnt)
  __builtin_amdgcn_s_wait_asynccnt(0);
#else
  asm volatile("s_wait_asynccnt 0x0" ::: "memory");
#endif
#endif
}

// ---- f32 -> bf16 bulk convert (one-shot, removes cvt from GEMM hot loops)
__global__ __launch_bounds__(256) void cvt_bf16(const float* __restrict__ in,
                                                bf16* __restrict__ out, int n4) {
  const int i = blockIdx.x * blockDim.x + threadIdx.x;
  if (i < n4) {
    float4 v = ((const float4*)in)[i];
    bf16x4 o;
    o[0] = (bf16)v.x; o[1] = (bf16)v.y; o[2] = (bf16)v.z; o[3] = (bf16)v.w;
    ((bf16x4*)out)[i] = o;
  }
}

// ---- projection GEMM: O[m,n] = sum_k X[m,k] * W[n,k] + bias[n] -----------
// X, W are bf16. Block tile 128x128, 8 waves (4x2), wave tile 32x64.
// K-step 64 (two WMMA K-slices per LDS stage).
// OMODE: 0 -> bf16 Q [B,H,S,D]   1 -> bf16 K [B,H,S,D]
//        2 -> bf16 Vt [B,H,D,S]  3 -> f32 [M,N] (d_out)
#define LDSTRIDE 72  // 64 + 8 pad elements; 144B row -> every 8-elem chunk 16B aligned

template <int OMODE>
__global__ __launch_bounds__(256) void proj_gemm(const bf16* __restrict__ X,
                                                 const bf16* __restrict__ W,
                                                 const float* __restrict__ bias,
                                                 bf16* __restrict__ outb,
                                                 float* __restrict__ outf) {
  __shared__ bf16 As[128 * LDSTRIDE];
  __shared__ bf16 Bs[128 * LDSTRIDE];

  const int t    = threadIdx.x;
  const int lane = t & 31;
  const int wave = t >> 5;
  const int waveM = wave & 3;   // 0..3 -> 32-row strip
  const int waveN = wave >> 2;  // 0..1 -> 64-col strip
  const int bm = blockIdx.y * 128;
  const int bn = blockIdx.x * 128;

  const int c8   = t & 7;   // which 8-element chunk inside the 64-wide K strip
  const int row0 = t >> 3;  // 0..31

  v8f acc[2][4] = {};

  const int hh  = lane >> 4;
  const int l16 = lane & 15;

  for (int k0 = 0; k0 < EMBED; k0 += 64) {
    __syncthreads();
    // ---- stage A (128x64) and B (128x64) bf16 tiles into LDS
#pragma unroll
    for (int rr = 0; rr < 128; rr += 32) {
      const int r = row0 + rr;
      stage16(X + (size_t)(bm + r) * EMBED + k0 + c8 * 8, As + r * LDSTRIDE + c8 * 8);
      stage16(W + (size_t)(bn + r) * EMBED + k0 + c8 * 8, Bs + r * LDSTRIDE + c8 * 8);
    }
    stage_wait();
    __syncthreads();

    // ---- two WMMA K-slices of 32
#pragma unroll
    for (int ks = 0; ks < 2; ++ks) {
      v16bf af[2], bfm[4];
#pragma unroll
      for (int mt = 0; mt < 2; ++mt) {
        const bf16* p = As + (waveM * 32 + mt * 16 + l16) * LDSTRIDE + ks * 32 + hh * 8;
        af[mt] = load_frag(p, p + 16);
      }
#pragma unroll
      for (int nt = 0; nt < 4; ++nt) {
        const bf16* p = Bs + (waveN * 64 + nt * 16 + l16) * LDSTRIDE + ks * 32 + hh * 8;
        bfm[nt] = load_frag(p, p + 16);
      }
#pragma unroll
      for (int mt = 0; mt < 2; ++mt)
#pragma unroll
        for (int nt = 0; nt < 4; ++nt)
          acc[mt][nt] = wmma_bf16(af[mt], bfm[nt], acc[mt][nt]);
    }
  }

  // ---- epilogue: bias add + layout-specific store
#pragma unroll
  for (int mt = 0; mt < 2; ++mt) {
#pragma unroll
    for (int nt = 0; nt < 4; ++nt) {
      const int n  = bn + waveN * 64 + nt * 16 + l16;
      const float bv = bias[n];
      const int mbase = bm + waveM * 32 + mt * 16 + hh * 8;
#pragma unroll
      for (int r = 0; r < 8; ++r) {
        const int m = mbase + r;
        const float v = acc[mt][nt][r] + bv;
        if constexpr (OMODE == 3) {
          outf[(size_t)m * EMBED + n] = v;
        } else {
          const int b = m >> 11, s = m & 2047;
          const int h = n >> 6,  d = n & 63;
          if constexpr (OMODE == 2) {
            // Vt: [B,H,D,S]
            outb[(((size_t)(b * HEADS + h) * HEADD + d) << 11) + s] = (bf16)v;
          } else {
            // Q/K: [B,H,S,D]
            outb[((((size_t)(b * HEADS + h) * SEQ) + s) << 6) + d] = (bf16)v;
          }
        }
      }
    }
  }
}

// ---- flash attention: one wave per 16-query tile -------------------------
// Transposed score tiles (S^T = K_blk x Q^T): softmax reductions are in-lane
// + one SWAPX16 swizzle, and P^T in C-layout maps directly onto the
// B-fragment layout of the ctx^T = V^T x P^T WMMA (zero cross-lane moves).
__global__ __launch_bounds__(128) void flash_attn(const bf16* __restrict__ Qh,
                                                  const bf16* __restrict__ Kh,
                                                  const bf16* __restrict__ Vt,
                                                  bf16* __restrict__ Ctx) {
  const int lane = threadIdx.x & 31;
  const int wave = threadIdx.x >> 5;
  const int tile = blockIdx.x * 4 + wave;  // 0..4095
  const int bh   = tile >> 7;              // 0..31  (b*16+h)
  const int qt   = tile & 127;             // query tile within head

  const int hh  = lane >> 4;
  const int l16 = lane & 15;

  // Q^T B-fragments (d = 0..31 and 32..63), loaded once
  const bf16* qbase = Qh + ((size_t)bh * SEQ + qt * 16 + l16) * HEADD;
  v16bf qf[2];
#pragma unroll
  for (int j = 0; j < 2; ++j) {
    const bf16* p = qbase + j * 32 + hh * 8;
    qf[j] = load_frag(p, p + 16);
  }

  v8f acc[4] = {};           // ctx^T tiles: rows d (16 each), col q
  float m_run = -1e30f;
  float s_run = 0.0f;

  for (int kb = 0; kb < SEQ / 32; ++kb) {
    // ---- score^T tiles: 16 keys x 16 queries, two per 32-key block
    v8f sc[2] = {};
#pragma unroll
    for (int tt = 0; tt < 2; ++tt) {
      const bf16* kr =
          Kh + ((size_t)bh * SEQ + kb * 32 + tt * 16 + l16) * HEADD + hh * 8;
#pragma unroll
      for (int j = 0; j < 2; ++j) {
        v16bf kf = load_frag(kr + j * 32, kr + j * 32 + 16);
        sc[tt] = wmma_bf16(kf, qf[j], sc[tt]);
      }
    }

    // ---- online softmax (per query column; lanes l and l+16 share a column)
    float pv[16];
#pragma unroll
    for (int r = 0; r < 8; ++r) {
      pv[r]     = sc[0][r] * SM_SCALE;
      pv[8 + r] = sc[1][r] * SM_SCALE;
    }
    float mloc = pv[0];
#pragma unroll
    for (int i = 1; i < 16; ++i) mloc = fmaxf(mloc, pv[i]);
    const float mblk = fmaxf(mloc, swz16(mloc));
    const float mnew = fmaxf(m_run, mblk);
    const float corr = __builtin_amdgcn_exp2f((m_run - mnew) * LOG2E);

    float e[16];
    float ssum = 0.0f;
#pragma unroll
    for (int i = 0; i < 16; ++i) {
      e[i] = __builtin_amdgcn_exp2f((pv[i] - mnew) * LOG2E);
      ssum += e[i];
    }
    const float blkSum = ssum + swz16(ssum);
    s_run = s_run * corr + blkSum;
    m_run = mnew;
#pragma unroll
    for (int i = 0; i < 4; ++i) acc[i] *= corr;

    // pack P^T into the B-fragment for K=32 (in-lane, no shuffles)
    v16bf pf;
#pragma unroll
    for (int i = 0; i < 16; ++i) pf[i] = (bf16)e[i];

    // ---- ctx^T += V^T x P^T (4 d-tiles)
#pragma unroll
    for (int i = 0; i < 4; ++i) {
      const bf16* vr =
          Vt + ((size_t)bh * HEADD + i * 16 + l16) * SEQ + kb * 32 + hh * 8;
      v16bf vf = load_frag(vr, vr + 16);
      acc[i] = wmma_bf16(vf, pf, acc[i]);
    }
  }

  // ---- normalize and store ctx[b, q, h*64 + d] as bf16
  const float inv = 1.0f / s_run;
  const int b    = bh >> 4;
  const int head = bh & 15;
  const int q    = qt * 16 + l16;
#pragma unroll
  for (int i = 0; i < 4; ++i) {
    bf16x8 ov;
#pragma unroll
    for (int r = 0; r < 8; ++r) ov[r] = (bf16)(acc[i][r] * inv);
    const size_t off =
        ((size_t)b * SEQ + q) * EMBED + head * HEADD + i * 16 + hh * 8;
    *(bf16x8*)(Ctx + off) = ov;
  }
}

// ---- launch --------------------------------------------------------------

extern "C" void kernel_launch(void* const* d_in, const int* in_sizes, int n_in,
                              void* d_out, int out_size, void* d_ws, size_t ws_size,
                              hipStream_t stream) {
  const float* query = (const float*)d_in[0];
  const float* key   = (const float*)d_in[1];
  const float* value = (const float*)d_in[2];
  const float* Wq = (const float*)d_in[3];
  const float* bq = (const float*)d_in[4];
  const float* Wk = (const float*)d_in[5];
  const float* bk = (const float*)d_in[6];
  const float* Wv = (const float*)d_in[7];
  const float* bv = (const float*)d_in[8];
  const float* Wo = (const float*)d_in[9];
  const float* bo = (const float*)d_in[10];

  const size_t NELT = (size_t)BATCH * HEADS * SEQ * HEADD;  // 4,194,304
  const size_t WELT = (size_t)EMBED * EMBED;                // 1,048,576
  bf16* Qh  = (bf16*)d_ws;          // attention operands / context
  bf16* Kh  = Qh + NELT;
  bf16* VtW = Kh + NELT;
  bf16* Ctx = VtW + NELT;
  bf16* Xq  = Ctx + NELT;           // bf16 copies of activations
  bf16* Xk  = Xq + NELT;
  bf16* Xv  = Xk + NELT;
  bf16* Wqb = Xv + NELT;            // bf16 copies of weights
  bf16* Wkb = Wqb + WELT;
  bf16* Wvb = Wkb + WELT;
  bf16* Wob = Wvb + WELT;

  // one-shot f32 -> bf16 conversion of activations and weights
  {
    const int n4x = (int)(NELT / 4);  // 1,048,576
    const int n4w = (int)(WELT / 4);  // 262,144
    cvt_bf16<<<(n4x + 255) / 256, 256, 0, stream>>>(query, Xq, n4x);
    cvt_bf16<<<(n4x + 255) / 256, 256, 0, stream>>>(key,   Xk, n4x);
    cvt_bf16<<<(n4x + 255) / 256, 256, 0, stream>>>(value, Xv, n4x);
    cvt_bf16<<<(n4w + 255) / 256, 256, 0, stream>>>(Wq, Wqb, n4w);
    cvt_bf16<<<(n4w + 255) / 256, 256, 0, stream>>>(Wk, Wkb, n4w);
    cvt_bf16<<<(n4w + 255) / 256, 256, 0, stream>>>(Wv, Wvb, n4w);
    cvt_bf16<<<(n4w + 255) / 256, 256, 0, stream>>>(Wo, Wob, n4w);
  }

  dim3 gp(EMBED / 128, MROWS / 128);  // (8, 32)
  dim3 bp(256);

  proj_gemm<0><<<gp, bp, 0, stream>>>(Xq, Wqb, bq, Qh, nullptr);
  proj_gemm<1><<<gp, bp, 0, stream>>>(Xk, Wkb, bk, Kh, nullptr);
  proj_gemm<2><<<gp, bp, 0, stream>>>(Xv, Wvb, bv, VtW, nullptr);

  // 4096 query tiles, 4 waves (128 threads) per block
  flash_attn<<<(BATCH * HEADS * (SEQ / 16)) / 4, 128, 0, stream>>>(Qh, Kh, VtW, Ctx);

  proj_gemm<3><<<gp, bp, 0, stream>>>(Ctx, Wob, bo, nullptr, (float*)d_out);
}
